// CnfProcessingBlock_59150289601135
// MI455X (gfx1250) — matrix operand order repaired and checked by reading, compiled, not run on hardware
//
#include <hip/hip_runtime.h>
#include <hip/hip_bf16.h>
#include <math.h>

#define N_NODES   100000
#define N_EDGES   600000
#define D         128
#define ED        16
#define NEG_SLOPE 0.2f

typedef __attribute__((ext_vector_type(16))) __bf16 v16bf;
typedef __attribute__((ext_vector_type(8)))  float  v8f;

// ---------------------------------------------------------------------------
// helpers
// ---------------------------------------------------------------------------
__device__ inline float lrelu(float v) { return v > 0.f ? v : NEG_SLOPE * v; }

// float atomic-max via signed/unsigned integer ordering trick (init = -inf)
__device__ inline void atomicMaxFloat(float* addr, float v) {
    if (v >= 0.f) atomicMax((int*)addr, __float_as_int(v));
    else          atomicMin((unsigned int*)addr, __float_as_uint(v));
}

// wave-per-node GEMV: out[d] = sum_k h[k] * W[k*D + d], 4 dims per lane
__device__ inline float4 gemv_row(const float* __restrict__ hrow,
                                  const float* __restrict__ W, int lane) {
    float4 acc = {0.f, 0.f, 0.f, 0.f};
    float4 hv = ((const float4*)hrow)[lane];           // k = lane*4 .. lane*4+3
#pragma unroll 4
    for (int j = 0; j < 32; ++j) {
        float h0 = __shfl(hv.x, j, 32);
        float h1 = __shfl(hv.y, j, 32);
        float h2 = __shfl(hv.z, j, 32);
        float h3 = __shfl(hv.w, j, 32);
        float4 w0 = ((const float4*)(W + (size_t)(4 * j + 0) * D))[lane];
        float4 w1 = ((const float4*)(W + (size_t)(4 * j + 1) * D))[lane];
        float4 w2 = ((const float4*)(W + (size_t)(4 * j + 2) * D))[lane];
        float4 w3 = ((const float4*)(W + (size_t)(4 * j + 3) * D))[lane];
        acc.x += h0 * w0.x + h1 * w1.x + h2 * w2.x + h3 * w3.x;
        acc.y += h0 * w0.y + h1 * w1.y + h2 * w2.y + h3 * w3.y;
        acc.z += h0 * w0.z + h1 * w1.z + h2 * w2.z + h3 * w3.z;
        acc.w += h0 * w0.w + h1 * w1.w + h2 * w2.w + h3 * w3.w;
    }
    return acc;
}

// ---------------------------------------------------------------------------
// 1) init workspace
// ---------------------------------------------------------------------------
__global__ void init_kernel(float* __restrict__ agg, float* __restrict__ segmax,
                            float* __restrict__ segsum) {
    int i = blockIdx.x * blockDim.x + threadIdx.x;
    if (i < N_NODES * D) agg[i] = 0.f;
    if (i < N_NODES) { segsum[i] = 0.f; segmax[i] = -INFINITY; }
}

// ---------------------------------------------------------------------------
// 2) XL[t] = bf16(h) @ bf16(Wl[t])   (WMMA, f32 accumulate)
//    block = 256 threads = 8 waves; each wave owns a 16(M) x 128(N) strip.
// ---------------------------------------------------------------------------
#define LDS_STRIDE 144   // 128 + 16 pad: keeps 16-elem k-segments 32B aligned, spreads banks

__global__ __launch_bounds__(256)
void xl_wmma_kernel(const float* __restrict__ h,
                    const float* __restrict__ Wl,   // [3, D, D]
                    float* __restrict__ XL) {       // [3, N, D]
    __shared__ __bf16 Wlds[D * LDS_STRIDE];
    const int t = blockIdx.y;
    const float* W = Wl + (size_t)t * D * D;

    // stage W transposed into LDS as bf16: Wlds[n*STRIDE + k] = W[k][n]
    for (int i = threadIdx.x; i < D * D; i += 256) {
        int k = i >> 7, n = i & 127;
        Wlds[n * LDS_STRIDE + k] = (__bf16)W[i];
    }
    __syncthreads();

    const int wave = threadIdx.x >> 5;
    const int lane = threadIdx.x & 31;
    const int half = lane >> 4;       // lane group (ISA A/B/C layouts)
    const int l16  = lane & 15;

    const int rowTile = blockIdx.x * 128 + wave * 16;
    int rowA = rowTile + l16;
    if (rowA >= N_NODES) rowA = N_NODES - 1;     // clamp loads; stores masked

    v8f acc[8] = {};

#pragma unroll
    for (int kt = 0; kt < 4; ++kt) {
        // A fragment (16x32, contiguous k, same order as B): elem j: k = kt*32 + half*16 + j
        const float* hrow = h + (size_t)rowA * D + kt * 32 + half * 16;
        float4 f0 = *(const float4*)(hrow + 0);
        float4 f1 = *(const float4*)(hrow + 4);
        float4 f2 = *(const float4*)(hrow + 8);
        float4 f3 = *(const float4*)(hrow + 12);
        v16bf a;
        a[0]  = (__bf16)f0.x; a[1]  = (__bf16)f0.y; a[2]  = (__bf16)f0.z; a[3]  = (__bf16)f0.w;
        a[4]  = (__bf16)f1.x; a[5]  = (__bf16)f1.y; a[6]  = (__bf16)f1.z; a[7]  = (__bf16)f1.w;
        a[8]  = (__bf16)f2.x; a[9]  = (__bf16)f2.y; a[10] = (__bf16)f2.z; a[11] = (__bf16)f2.w;
        a[12] = (__bf16)f3.x; a[13] = (__bf16)f3.y; a[14] = (__bf16)f3.z; a[15] = (__bf16)f3.w;

        // B fragment (32x16): lane column n = nt*16+l16; elem j: k = kt*32 + half*16 + j
        const int kbase = kt * 32 + half * 16;
#pragma unroll
        for (int nt = 0; nt < 8; ++nt) {
            const int ncol = nt * 16 + l16;
            v16bf b = *(const v16bf*)(&Wlds[ncol * LDS_STRIDE + kbase]);
            acc[nt] = __builtin_amdgcn_wmma_f32_16x16x32_bf16(
                false, a, false, b, (short)0, acc[nt], false, false);
        }
    }

    // C/D layout: lane n = l16, VGPR r -> M = r + half*8
    float* outp = XL + (size_t)t * N_NODES * D;
#pragma unroll
    for (int nt = 0; nt < 8; ++nt) {
#pragma unroll
        for (int r = 0; r < 8; ++r) {
            int m = rowTile + r + half * 8;
            if (m < N_NODES) outp[(size_t)m * D + nt * 16 + l16] = acc[nt][r];
        }
    }
}

// ---------------------------------------------------------------------------
// 3) XRsel[n] = h[n] @ Wr[node_type[n]]
// ---------------------------------------------------------------------------
__global__ __launch_bounds__(256)
void xr_kernel(const float* __restrict__ h, const int* __restrict__ ntype,
               const float* __restrict__ Wr, float* __restrict__ XR) {
    int n = blockIdx.x * 8 + (threadIdx.x >> 5);
    if (n >= N_NODES) return;
    int lane = threadIdx.x & 31;
    int t = ntype[n];
    float4 r = gemv_row(h + (size_t)n * D, Wr + (size_t)t * D * D, lane);
    ((float4*)(XR + (size_t)n * D))[lane] = r;
}

// ---------------------------------------------------------------------------
// 4) per-edge logit + segment max (wave per edge, t = type of dst)
// ---------------------------------------------------------------------------
__global__ __launch_bounds__(256)
void edge_logit_kernel(const float* __restrict__ XL, const float* __restrict__ XR,
                       const float* __restrict__ edge_attr,
                       const int* __restrict__ src, const int* __restrict__ dst,
                       const int* __restrict__ ntype,
                       const float* __restrict__ We, const float* __restrict__ att,
                       float* __restrict__ logit, float* __restrict__ segmax) {
    int e = blockIdx.x * 8 + (threadIdx.x >> 5);
    if (e >= N_EDGES) return;
    int lane = threadIdx.x & 31;
    int s = src[e], d = dst[e];
    int t = ntype[d];

    float4 xl = ((const float4*)(XL + ((size_t)t * N_NODES + s) * D))[lane];
    float4 xr = ((const float4*)(XR + (size_t)d * D))[lane];

    // xe = edge_attr[e] @ We[t]
    float ea = (lane < ED) ? edge_attr[(size_t)e * ED + lane] : 0.f;
    float4 xe = {0.f, 0.f, 0.f, 0.f};
    const float* Wt = We + (size_t)t * ED * D;
#pragma unroll
    for (int j = 0; j < ED; ++j) {
        float a = __shfl(ea, j, 32);
        float4 w = ((const float4*)(Wt + (size_t)j * D))[lane];
        xe.x += a * w.x; xe.y += a * w.y; xe.z += a * w.z; xe.w += a * w.w;
    }

    float4 z;
    z.x = lrelu(xl.x + xr.x + xe.x);
    z.y = lrelu(xl.y + xr.y + xe.y);
    z.z = lrelu(xl.z + xr.z + xe.z);
    z.w = lrelu(xl.w + xr.w + xe.w);

    float4 at = ((const float4*)(att + (size_t)t * D))[lane];
    float p = z.x * at.x + z.y * at.y + z.z * at.z + z.w * at.w;
#pragma unroll
    for (int off = 16; off > 0; off >>= 1) p += __shfl_xor(p, off, 32);

    if (lane == 0) {
        logit[e] = p;
        atomicMaxFloat(&segmax[d], p);
    }
}

// ---------------------------------------------------------------------------
// 5) exp(logit - max) + segment sum (thread per edge)
// ---------------------------------------------------------------------------
__global__ void edge_exp_kernel(float* __restrict__ logit, const int* __restrict__ dst,
                                const float* __restrict__ segmax,
                                float* __restrict__ segsum) {
    int e = blockIdx.x * blockDim.x + threadIdx.x;
    if (e >= N_EDGES) return;
    int d = dst[e];
    float ex = __expf(logit[e] - segmax[d]);
    logit[e] = ex;
    atomicAdd(&segsum[d], ex);
}

// ---------------------------------------------------------------------------
// 6) scatter alpha * XL[t][src] into AGG[dst] (wave per edge)
// ---------------------------------------------------------------------------
__global__ __launch_bounds__(256)
void edge_scatter_kernel(const float* __restrict__ XL, const float* __restrict__ logit,
                         const float* __restrict__ segsum,
                         const int* __restrict__ src, const int* __restrict__ dst,
                         const int* __restrict__ ntype, float* __restrict__ agg) {
    int e = blockIdx.x * 8 + (threadIdx.x >> 5);
    if (e >= N_EDGES) return;
    int lane = threadIdx.x & 31;
    int s = src[e], d = dst[e];
    int t = ntype[d];
    float alpha = logit[e] / segsum[d];
    float4 xl = ((const float4*)(XL + ((size_t)t * N_NODES + s) * D))[lane];
    float* ap = agg + (size_t)d * D + lane * 4;
    atomicAdd(ap + 0, alpha * xl.x);
    atomicAdd(ap + 1, alpha * xl.y);
    atomicAdd(ap + 2, alpha * xl.z);
    atomicAdd(ap + 3, alpha * xl.w);
}

// ---------------------------------------------------------------------------
// 7) out = relu(AGG + h @ Wres[t] + bias[t])  (wave per node)
// ---------------------------------------------------------------------------
__global__ __launch_bounds__(256)
void final_kernel(const float* __restrict__ h, const int* __restrict__ ntype,
                  const float* __restrict__ Wres, const float* __restrict__ bias,
                  const float* __restrict__ agg, float* __restrict__ out) {
    int n = blockIdx.x * 8 + (threadIdx.x >> 5);
    if (n >= N_NODES) return;
    int lane = threadIdx.x & 31;
    int t = ntype[n];
    float4 r = gemv_row(h + (size_t)n * D, Wres + (size_t)t * D * D, lane);
    float4 g = ((const float4*)(agg + (size_t)n * D))[lane];
    float4 b = ((const float4*)(bias + (size_t)t * D))[lane];
    float4 o;
    o.x = fmaxf(0.f, g.x + r.x + b.x);
    o.y = fmaxf(0.f, g.y + r.y + b.y);
    o.z = fmaxf(0.f, g.z + r.z + b.z);
    o.w = fmaxf(0.f, g.w + r.w + b.w);
    ((float4*)(out + (size_t)n * D))[lane] = o;
}

// ---------------------------------------------------------------------------
extern "C" void kernel_launch(void* const* d_in, const int* in_sizes, int n_in,
                              void* d_out, int out_size, void* d_ws, size_t ws_size,
                              hipStream_t stream) {
    const float* h         = (const float*)d_in[0];
    const int*   edge_idx  = (const int*)  d_in[1];   // [2, E]
    const float* edge_attr = (const float*)d_in[2];   // [E, 16]
    const int*   ntype     = (const int*)  d_in[3];   // [N]
    const float* Wl        = (const float*)d_in[4];   // [3, D, D]
    const float* Wr        = (const float*)d_in[5];   // [3, D, D]
    const float* We        = (const float*)d_in[6];   // [3, 16, D]
    const float* att       = (const float*)d_in[7];   // [3, D]
    const float* Wres      = (const float*)d_in[8];   // [3, D, D]
    const float* bias      = (const float*)d_in[9];   // [3, D]
    float* out = (float*)d_out;

    const int* src = edge_idx;
    const int* dst = edge_idx + N_EDGES;

    // workspace layout (floats)
    float* ws     = (float*)d_ws;
    float* XL     = ws;                                   // 3*N*D
    float* XR     = XL + (size_t)3 * N_NODES * D;         // N*D
    float* AGG    = XR + (size_t)N_NODES * D;             // N*D
    float* LOGIT  = AGG + (size_t)N_NODES * D;            // E
    float* SEGMAX = LOGIT + N_EDGES;                      // N
    float* SEGSUM = SEGMAX + N_NODES;                     // N

    init_kernel<<<(N_NODES * D + 255) / 256, 256, 0, stream>>>(AGG, SEGMAX, SEGSUM);

    dim3 g1((N_NODES + 127) / 128, 3);
    xl_wmma_kernel<<<g1, 256, 0, stream>>>(h, Wl, XL);

    xr_kernel<<<(N_NODES + 7) / 8, 256, 0, stream>>>(h, ntype, Wr, XR);

    edge_logit_kernel<<<(N_EDGES + 7) / 8, 256, 0, stream>>>(
        XL, XR, edge_attr, src, dst, ntype, We, att, LOGIT, SEGMAX);

    edge_exp_kernel<<<(N_EDGES + 255) / 256, 256, 0, stream>>>(LOGIT, dst, SEGMAX, SEGSUM);

    edge_scatter_kernel<<<(N_EDGES + 7) / 8, 256, 0, stream>>>(
        XL, LOGIT, SEGSUM, src, dst, ntype, AGG);

    final_kernel<<<(N_NODES + 7) / 8, 256, 0, stream>>>(h, ntype, Wres, bias, AGG, out);
}
